// MultiheadLatentAttention_17755394801798
// MI455X (gfx1250) — matrix operand order repaired and checked by reading, not compiled, not run on hardware
//
#include <hip/hip_runtime.h>
#include <math.h>

typedef __bf16 bf16_t;
typedef __attribute__((ext_vector_type(16))) __bf16 v16bf;
typedef __attribute__((ext_vector_type(8)))  __bf16 v8bf;
typedef __attribute__((ext_vector_type(8)))  float  v8f;

#define B_   2
#define L_   1024
#define DM   1024
#define DCKV 256
#define DCQ  512
#define NH   8
#define DH   64
#define DR   32
#define NIH  4
#define KTS  96
#define LWIN 64

// ---------- helpers ----------
static __device__ __forceinline__ bf16_t f2bf(float f) {
  unsigned u = __float_as_uint(f);
  unsigned lsb = (u >> 16) & 1u;
  u += 0x7fffu + lsb;
  union { unsigned short s; bf16_t h; } cv;
  cv.s = (unsigned short)(u >> 16);
  return cv.h;
}
static __device__ __forceinline__ float softplusf(float x) {
  return (x > 20.f) ? x : log1pf(expf(x));
}

union UA { v16bf v; v8bf h[2]; };

// ---------- conversion kernels ----------
__global__ void k_to_bf16(const float* __restrict__ in, bf16_t* __restrict__ out, int n) {
  int i = blockIdx.x * 256 + threadIdx.x;
  if (i < n) out[i] = f2bf(in[i]);
}

// in: fp32 [rows(K) x cols(N)] row-major  ->  out: bf16 [N x K] row-major (ld = rows)
__global__ void k_transpose_bf16(const float* __restrict__ in, bf16_t* __restrict__ out,
                                 int rows, int cols) {
  int i = blockIdx.x * 256 + threadIdx.x;
  if (i < rows * cols) {
    int r = i / cols, c = i % cols;
    out[(size_t)c * rows + r] = f2bf(in[i]);
  }
}

// ---------- register-blocked bf16 WMMA GEMM: C[M,N] = A[M,K] * B^T (B stored [N,K]) ----
// Wave tile = (MW*16) x (NW*16); block = 8 waves arranged 4(M) x 2(N).
// Each A fragment is reused NW times, each B fragment MW times -> fewer loads per WMMA.
template <int MW, int NW>
__global__ __launch_bounds__(256) void k_gemm_bf16(
    const bf16_t* __restrict__ A, long long sA, int lda,
    const bf16_t* __restrict__ Bm, long long sB, int ldb,
    float* __restrict__ outF, bf16_t* __restrict__ outH, long long sC, int ldc,
    int K) {
  int lane = threadIdx.x & 31;
  int wave = threadIdx.x >> 5;
  int z = blockIdx.z;
  const bf16_t* Ab = A + (size_t)z * sA;
  const bf16_t* Bb = Bm + (size_t)z * sB;
  int m0 = (blockIdx.y * 4 + (wave & 3)) * (MW * 16);
  int n0 = (blockIdx.x * 2 + (wave >> 2)) * (NW * 16);
  int l15 = lane & 15;
  int aklo = (lane < 16) ? 0 : 8;   // ISA 16-bit A layout: two groups of 8 consecutive K
  int bklo = (lane < 16) ? 0 : 8;   // B layout matches A: two groups of 8 consecutive K

  const bf16_t* ap[MW];
#pragma unroll
  for (int mi = 0; mi < MW; ++mi)
    ap[mi] = Ab + (size_t)(m0 + mi * 16 + l15) * lda + aklo;
  const bf16_t* bp[NW];
#pragma unroll
  for (int ni = 0; ni < NW; ++ni)
    bp[ni] = Bb + (size_t)(n0 + ni * 16 + l15) * ldb + bklo;

  v8f acc[MW][NW];
#pragma unroll
  for (int mi = 0; mi < MW; ++mi)
#pragma unroll
    for (int ni = 0; ni < NW; ++ni) acc[mi][ni] = (v8f){};

  for (int k0 = 0; k0 < K; k0 += 32) {
    UA a[MW], b[NW];
#pragma unroll
    for (int mi = 0; mi < MW; ++mi) {
      a[mi].h[0] = *(const v8bf*)(ap[mi] + k0);
      a[mi].h[1] = *(const v8bf*)(ap[mi] + k0 + 16);
    }
#pragma unroll
    for (int ni = 0; ni < NW; ++ni) {
      b[ni].h[0] = *(const v8bf*)(bp[ni] + k0);
      b[ni].h[1] = *(const v8bf*)(bp[ni] + k0 + 16);
    }
#pragma unroll
    for (int mi = 0; mi < MW; ++mi)
#pragma unroll
      for (int ni = 0; ni < NW; ++ni)
        acc[mi][ni] = __builtin_amdgcn_wmma_f32_16x16x32_bf16(
            false, a[mi].v, false, b[ni].v, (short)0, acc[mi][ni], false, false);
  }

#pragma unroll
  for (int mi = 0; mi < MW; ++mi) {
    int crow = m0 + mi * 16 + ((lane < 16) ? 0 : 8);
#pragma unroll
    for (int ni = 0; ni < NW; ++ni) {
      int ccol = n0 + ni * 16 + l15;
      if (outF) {
        float* o = outF + (size_t)z * sC + (size_t)crow * ldc + ccol;
#pragma unroll
        for (int r = 0; r < 8; ++r) o[(size_t)r * ldc] = acc[mi][ni][r];
      }
      if (outH) {
        bf16_t* o = outH + (size_t)z * sC + (size_t)crow * ldc + ccol;
#pragma unroll
        for (int r = 0; r < 8; ++r) o[(size_t)r * ldc] = f2bf(acc[mi][ni][r]);
      }
    }
  }
}

// ---------- indexer GEMM (2x2 blocked) with relu+head-sum epilogue ----------
// A = q_i [4096 x 64] per batch (rows = t*4+h), B = k_i [1024 x 64] per batch.
// Epilogue: I[b, t, s] = sum_h idx_w[h] * relu(dot).  4 consecutive C-rows = one token.
__global__ __launch_bounds__(256) void k_gemm_indexer(
    const bf16_t* __restrict__ A, long long sA,
    const bf16_t* __restrict__ Bm, long long sB,
    float* __restrict__ Iout, long long sI,
    const float* __restrict__ idx_w, int K) {
  int lane = threadIdx.x & 31;
  int wave = threadIdx.x >> 5;
  int z = blockIdx.z;
  const bf16_t* Ab = A + (size_t)z * sA;
  const bf16_t* Bb = Bm + (size_t)z * sB;
  int m0 = (blockIdx.y * 4 + (wave & 3)) * 32;
  int n0 = (blockIdx.x * 2 + (wave >> 2)) * 32;
  int l15 = lane & 15;
  int aklo = (lane < 16) ? 0 : 8;
  int bklo = (lane < 16) ? 0 : 8;

  const bf16_t* ap[2];
  const bf16_t* bp[2];
#pragma unroll
  for (int i = 0; i < 2; ++i) {
    ap[i] = Ab + (size_t)(m0 + i * 16 + l15) * 64 + aklo;
    bp[i] = Bb + (size_t)(n0 + i * 16 + l15) * 64 + bklo;
  }

  v8f acc[2][2];
#pragma unroll
  for (int mi = 0; mi < 2; ++mi)
#pragma unroll
    for (int ni = 0; ni < 2; ++ni) acc[mi][ni] = (v8f){};

  for (int k0 = 0; k0 < K; k0 += 32) {
    UA a[2], b[2];
#pragma unroll
    for (int i = 0; i < 2; ++i) {
      a[i].h[0] = *(const v8bf*)(ap[i] + k0);
      a[i].h[1] = *(const v8bf*)(ap[i] + k0 + 16);
      b[i].h[0] = *(const v8bf*)(bp[i] + k0);
      b[i].h[1] = *(const v8bf*)(bp[i] + k0 + 16);
    }
#pragma unroll
    for (int mi = 0; mi < 2; ++mi)
#pragma unroll
      for (int ni = 0; ni < 2; ++ni)
        acc[mi][ni] = __builtin_amdgcn_wmma_f32_16x16x32_bf16(
            false, a[mi].v, false, b[ni].v, (short)0, acc[mi][ni], false, false);
  }

  float w0 = idx_w[0], w1 = idx_w[1], w2 = idx_w[2], w3 = idx_w[3];
  float* Ob = Iout + (size_t)z * sI;
#pragma unroll
  for (int mi = 0; mi < 2; ++mi) {
    int crow = m0 + mi * 16 + ((lane < 16) ? 0 : 8);  // multiple of 8 -> token aligned
    int tok0 = crow >> 2;
#pragma unroll
    for (int ni = 0; ni < 2; ++ni) {
      int ccol = n0 + ni * 16 + l15;
      v8f c = acc[mi][ni];
      float i0 = w0 * fmaxf(c[0], 0.f) + w1 * fmaxf(c[1], 0.f)
               + w2 * fmaxf(c[2], 0.f) + w3 * fmaxf(c[3], 0.f);
      float i1 = w0 * fmaxf(c[4], 0.f) + w1 * fmaxf(c[5], 0.f)
               + w2 * fmaxf(c[6], 0.f) + w3 * fmaxf(c[7], 0.f);
      Ob[(size_t)tok0 * L_ + ccol]       = i0;
      Ob[(size_t)(tok0 + 1) * L_ + ccol] = i1;
    }
  }
}

// ---------- top-k (96 of 1024) per query row; replicates jax top_k tie-breaking ----------
__global__ __launch_bounds__(256) void k_topk(const float* __restrict__ I,
                                              int* __restrict__ out) {
  __shared__ unsigned keys[L_];
  __shared__ unsigned long long red[256];
  int tid = threadIdx.x;
  int bt = blockIdx.x;
  int t = bt & (L_ - 1);
  for (int s = tid; s < L_; s += 256) {
    unsigned key;
    if (s > t) {
      key = 0x007FFFFFu;                       // -inf (causal), ranked below all finite
    } else if (s >= t - (LWIN - 1)) {
      key = 0xFF800000u;                       // +inf (forced local window)
    } else {
      unsigned ub = __float_as_uint(I[(size_t)bt * L_ + s]);
      key = (ub & 0x80000000u) ? ~ub : (ub | 0x80000000u);  // order-preserving
    }
    keys[s] = key;
  }
  __syncthreads();
  for (int it = 0; it < KTS; ++it) {
    unsigned long long best = 0ull;
    for (int s = tid; s < L_; s += 256) {
      unsigned long long comp =
          ((unsigned long long)keys[s] << 32) | (unsigned)(~s);  // ties -> smallest s
      if (comp > best) best = comp;
    }
    red[tid] = best;
    __syncthreads();
    for (int o = 128; o > 0; o >>= 1) {
      if (tid < o) { if (red[tid + o] > red[tid]) red[tid] = red[tid + o]; }
      __syncthreads();
    }
    if (tid == 0) {
      int sel = (int)(~(unsigned)red[0]);
      out[(size_t)bt * KTS + it] = sel;
      keys[sel] = 0u;                          // remove (below every live key)
    }
    __syncthreads();
  }
}

// ---------- fused sparse attention per (b,t) ----------
__global__ __launch_bounds__(256) void k_attn(
    const float* __restrict__ qc, const float* __restrict__ qr,
    const float* __restrict__ kup, const float* __restrict__ vup,
    const float* __restrict__ kr, const int* __restrict__ topk,
    const float* __restrict__ raw_delta, bf16_t* __restrict__ attn_out) {
  __shared__ int   s_idx[KTS];
  __shared__ float s_qc[NH * DH];
  __shared__ float s_qr[NH * DR];
  __shared__ float s_kr[KTS * DR];
  __shared__ float s_sc[NH * KTS];
  int tid = threadIdx.x;
  int bt = blockIdx.x;
  int b = bt >> 10, t = bt & (L_ - 1);

  if (tid < KTS) s_idx[tid] = topk[(size_t)bt * KTS + tid];
  for (int i = tid; i < NH * DH; i += 256) s_qc[i] = qc[(size_t)bt * (NH * DH) + i];

  if (tid < NH * 16) {                        // query PoPE (angles from t)
    int h = tid >> 4, i = tid & 15;
    float th = powf(10000.f, -(float)i / 16.f);
    float dl = -6.283185307179586f / (1.f + expf(-raw_delta[i]));
    float ang = (float)t * th + dl;
    float c = cosf(ang), s = sinf(ang);
    float x1 = qr[(size_t)bt * (NH * DR) + h * DR + i];
    float x2 = qr[(size_t)bt * (NH * DR) + h * DR + 16 + i];
    float m1 = softplusf(x1), m2 = softplusf(x2);
    s_qr[h * DR + i]      = m1 * c - m2 * s;
    s_qr[h * DR + 16 + i] = m1 * s + m2 * c;
  }
  __syncthreads();

  for (int wi = tid; wi < KTS * 16; wi += 256) {  // key PoPE (angles from slot index)
    int k = wi >> 4, i = wi & 15;
    int row = (b << 10) + s_idx[k];
    float th = powf(10000.f, -(float)i / 16.f);
    float dl = -6.283185307179586f / (1.f + expf(-raw_delta[i]));
    float ang = (float)k * th + dl;
    float c = cosf(ang), s = sinf(ang);
    float x1 = kr[(size_t)row * DR + i];
    float x2 = kr[(size_t)row * DR + 16 + i];
    float m1 = softplusf(x1), m2 = softplusf(x2);
    s_kr[k * DR + i]      = m1 * c - m2 * s;
    s_kr[k * DR + 16 + i] = m1 * s + m2 * c;
  }
  __syncthreads();

  const float scale = 0.10206207261596577f;   // (64+32)^-0.5
  for (int wi = tid; wi < NH * KTS; wi += 256) {
    int h = wi / KTS, k = wi % KTS;
    int row = (b << 10) + s_idx[k];
    const float* kp = kup + (size_t)row * (NH * DH) + h * DH;
    float acc = 0.f;
    for (int d = 0; d < DH; ++d) acc += s_qc[h * DH + d] * kp[d];
    const float* krp = &s_kr[k * DR];
    for (int i = 0; i < DR; ++i) acc += s_qr[h * DR + i] * krp[i];
    s_sc[h * KTS + k] = acc * scale;
  }
  __syncthreads();

  {                                           // softmax: one wave32 per head
    int w = tid >> 5, lane = tid & 31;
    float v0 = s_sc[w * KTS + lane];
    float v1 = s_sc[w * KTS + lane + 32];
    float v2 = s_sc[w * KTS + lane + 64];
    float m = fmaxf(v0, fmaxf(v1, v2));
    for (int o = 16; o; o >>= 1) m = fmaxf(m, __shfl_xor(m, o, 32));
    float e0 = expf(v0 - m), e1 = expf(v1 - m), e2 = expf(v2 - m);
    float ssum = e0 + e1 + e2;
    for (int o = 16; o; o >>= 1) ssum += __shfl_xor(ssum, o, 32);
    float inv = 1.f / ssum;
    s_sc[w * KTS + lane]      = e0 * inv;
    s_sc[w * KTS + lane + 32] = e1 * inv;
    s_sc[w * KTS + lane + 64] = e2 * inv;
  }
  __syncthreads();

  for (int wi = tid; wi < NH * DH; wi += 256) {
    int h = wi >> 6, d = wi & 63;
    float acc = 0.f;
    for (int k = 0; k < KTS; ++k) {
      int row = (b << 10) + s_idx[k];
      acc += s_sc[h * KTS + k] * vup[(size_t)row * (NH * DH) + h * DH + d];
    }
    attn_out[(size_t)bt * (NH * DH) + wi] = f2bf(acc);
  }
}

// ---------- host launcher ----------
extern "C" void kernel_launch(void* const* d_in, const int* in_sizes, int n_in,
                              void* d_out, int out_size, void* d_ws, size_t ws_size,
                              hipStream_t stream) {
  (void)in_sizes; (void)n_in; (void)out_size; (void)ws_size;
  const float* x      = (const float*)d_in[0];
  const float* W_dkv  = (const float*)d_in[1];
  const float* W_uk   = (const float*)d_in[2];
  const float* W_uv   = (const float*)d_in[3];
  const float* W_dq   = (const float*)d_in[4];
  const float* W_uq   = (const float*)d_in[5];
  const float* W_qr   = (const float*)d_in[6];
  const float* W_kr   = (const float*)d_in[7];
  const float* W_out  = (const float*)d_in[8];
  const float* idx_wq = (const float*)d_in[9];
  const float* idx_wk = (const float*)d_in[10];
  const float* idx_w  = (const float*)d_in[11];
  const float* rdelta = (const float*)d_in[12];

  const int BL = B_ * L_;  // 2048
  size_t off = 0;
  auto alloc = [&](size_t bytes) -> void* {
    void* p = (char*)d_ws + off;
    off += (bytes + 255) & ~(size_t)255;
    return p;
  };
  bf16_t* xh    = (bf16_t*)alloc((size_t)BL * DM * 2);
  bf16_t* wdkvT = (bf16_t*)alloc((size_t)DM * DCKV * 2);
  bf16_t* wukT  = (bf16_t*)alloc((size_t)DCKV * NH * DH * 2);
  bf16_t* wuvT  = (bf16_t*)alloc((size_t)DCKV * NH * DH * 2);
  bf16_t* wdqT  = (bf16_t*)alloc((size_t)DM * DCQ * 2);
  bf16_t* wuqT  = (bf16_t*)alloc((size_t)DCQ * NH * DH * 2);
  bf16_t* wqrT  = (bf16_t*)alloc((size_t)DCQ * NH * DR * 2);
  bf16_t* wkrT  = (bf16_t*)alloc((size_t)DM * DR * 2);
  bf16_t* woutT = (bf16_t*)alloc((size_t)NH * DH * DM * 2);
  bf16_t* wiqT  = (bf16_t*)alloc((size_t)DM * NIH * DH * 2);
  bf16_t* wikT  = (bf16_t*)alloc((size_t)DM * DH * 2);
  bf16_t* ckvh  = (bf16_t*)alloc((size_t)BL * DCKV * 2);
  bf16_t* cqh   = (bf16_t*)alloc((size_t)BL * DCQ * 2);
  float*  kupf  = (float*)alloc((size_t)BL * NH * DH * 4);
  float*  vupf  = (float*)alloc((size_t)BL * NH * DH * 4);
  float*  qcf   = (float*)alloc((size_t)BL * NH * DH * 4);
  float*  qrf   = (float*)alloc((size_t)BL * NH * DR * 4);
  float*  krf   = (float*)alloc((size_t)BL * DR * 4);
  bf16_t* qih   = (bf16_t*)alloc((size_t)BL * NIH * DH * 2);
  bf16_t* kih   = (bf16_t*)alloc((size_t)BL * DH * 2);
  float*  If    = (float*)alloc((size_t)B_ * L_ * L_ * 4);
  int*    tk    = (int*)alloc((size_t)B_ * L_ * KTS * 4);
  bf16_t* atth  = (bf16_t*)alloc((size_t)BL * NH * DH * 2);

  // 1) conversions
  {
    int n = BL * DM;
    k_to_bf16<<<(n + 255) / 256, 256, 0, stream>>>(x, xh, n);
  }
  auto T = [&](const float* in, bf16_t* out, int rows, int cols) {
    int n = rows * cols;
    k_transpose_bf16<<<(n + 255) / 256, 256, 0, stream>>>(in, out, rows, cols);
  };
  T(W_dkv,  wdkvT, DM,   DCKV);
  T(W_uk,   wukT,  DCKV, NH * DH);
  T(W_uv,   wuvT,  DCKV, NH * DH);
  T(W_dq,   wdqT,  DM,   DCQ);
  T(W_uq,   wuqT,  DCQ,  NH * DH);
  T(W_qr,   wqrT,  DCQ,  NH * DR);
  T(W_kr,   wkrT,  DM,   DR);
  T(W_out,  woutT, NH * DH, DM);
  T(idx_wq, wiqT,  DM,   NIH * DH);
  T(idx_wk, wikT,  DM,   DH);

  // 2) dense WMMA GEMMs (wave tile 32x32 except the narrow N=32 case)
  auto G22 = [&](const bf16_t* A, int lda, const bf16_t* Bm, int ldb,
                 float* oF, bf16_t* oH, int ldc, int M, int N, int K) {
    dim3 grid(N / 64, M / 128, 1);
    k_gemm_bf16<2, 2><<<grid, 256, 0, stream>>>(A, 0, lda, Bm, 0, ldb, oF, oH, 0, ldc, K);
  };
  auto G21 = [&](const bf16_t* A, int lda, const bf16_t* Bm, int ldb,
                 float* oF, bf16_t* oH, int ldc, int M, int N, int K) {
    dim3 grid(N / 32, M / 128, 1);
    k_gemm_bf16<2, 1><<<grid, 256, 0, stream>>>(A, 0, lda, Bm, 0, ldb, oF, oH, 0, ldc, K);
  };
  G22(xh,   DM,   wdkvT, DM,   nullptr, ckvh, DCKV,    BL, DCKV,    DM);   // c_kv
  G22(ckvh, DCKV, wukT,  DCKV, kupf,    nullptr, NH*DH, BL, NH*DH, DCKV);  // k up-proj
  G22(ckvh, DCKV, wuvT,  DCKV, vupf,    nullptr, NH*DH, BL, NH*DH, DCKV);  // v up-proj
  G22(xh,   DM,   wdqT,  DM,   nullptr, cqh,  DCQ,     BL, DCQ,     DM);   // c_q
  G22(cqh,  DCQ,  wuqT,  DCQ,  qcf,     nullptr, NH*DH, BL, NH*DH, DCQ);   // query_c
  G22(cqh,  DCQ,  wqrT,  DCQ,  qrf,     nullptr, NH*DR, BL, NH*DR, DCQ);   // query_r pre-PoPE
  G21(xh,   DM,   wkrT,  DM,   krf,     nullptr, DR,    BL, DR,     DM);   // k_rope pre-gather
  G22(xh,   DM,   wiqT,  DM,   nullptr, qih,  NIH*DH,   BL, NIH*DH, DM);   // indexer q
  G22(xh,   DM,   wikT,  DM,   nullptr, kih,  DH,       BL, DH,     DM);   // indexer k

  // 3) indexer scores I[b,t,s] (2x2 WMMA + relu/head-sum epilogue), batched over z
  {
    dim3 grid(L_ / 64, (L_ * NIH) / 128, B_);
    k_gemm_indexer<<<grid, 256, 0, stream>>>(
        qih, (long long)L_ * NIH * DH, kih, (long long)L_ * DH,
        If, (long long)L_ * L_, idx_w, DH);
  }

  // 4) token selection
  k_topk<<<BL, 256, 0, stream>>>(If, tk);

  // 5) fused sparse attention (gather + PoPE + softmax + V)
  k_attn<<<BL, 256, 0, stream>>>(qcf, qrf, kupf, vupf, krf, tk, rdelta, atth);

  // 6) output projection straight into d_out (fp32)
  G22(atth, NH * DH, woutT, NH * DH, (float*)d_out, nullptr, DM, BL, DM, NH * DH);
}
